// LFC_11175504904474
// MI455X (gfx1250) — 8-bit matrix operand order repaired and checked by reading, compiled, not run on hardware
//
#include <hip/hip_runtime.h>
#include <stdint.h>


typedef __attribute__((ext_vector_type(16))) int   v16i;
typedef __attribute__((ext_vector_type(8)))  float v8f;
typedef __attribute__((ext_vector_type(4)))  int   v4i;
typedef __attribute__((ext_vector_type(2)))  int   v2i;

#define AS1 __attribute__((address_space(1)))
#define AS3 __attribute__((address_space(3)))

#define BATCH 8192
#define HID   1024
#define DIN   784
#define KP1   896
#define NCLS  10

#if __has_builtin(__builtin_amdgcn_global_load_async_to_lds_b128)
#define USE_ASYNC_LDS 1
#else
#define USE_ASYNC_LDS 0
#endif

static __device__ __forceinline__ void wait_asynccnt0() {
#if __has_builtin(__builtin_amdgcn_s_wait_asynccnt)
  __builtin_amdgcn_s_wait_asynccnt(0);
#else
  asm volatile("s_wait_asynccnt 0x0" ::: "memory");
#endif
}

__global__ void bnn_quant_input(const float* __restrict__ x, uint8_t* __restrict__ a0) {
  int idx = blockIdx.x * 256 + threadIdx.x;
  int row = idx / KP1;
  int col = idx - row * KP1;
  uint8_t b = 0;
  if (col < DIN) {
    float t = 2.0f * x[(size_t)row * DIN + col] - 1.0f;
    b = (t >= 0.0f) ? 0x38 : 0xB8;
  }
  a0[idx] = b;
}

__global__ void bnn_quant_weight(const float* __restrict__ w, uint8_t* __restrict__ wq,
                                 float* __restrict__ scale, int K, int Kp) {
  int row = blockIdx.x;
  int tid = threadIdx.x;
  __shared__ float red[256];
  float s = 0.0f;
  for (int k = tid; k < K; k += 256) s += fabsf(w[(size_t)row * K + k]);
  red[tid] = s; __syncthreads();
  for (int off = 128; off > 0; off >>= 1) {
    if (tid < off) red[tid] += red[tid + off];
    __syncthreads();
  }
  if (tid == 0) scale[row] = red[0] / (float)K;
  for (int k = tid; k < Kp; k += 256) {
    uint8_t b = 0;
    if (k < K) b = (w[(size_t)row * K + k] >= 0.0f) ? 0x38 : 0xB8;
    wq[(size_t)row * Kp + k] = b;
  }
}

__global__ void bnn_quant_fc(const float* __restrict__ wfc, uint8_t* __restrict__ wqs,
                             float* __restrict__ scale_fc) {
  int tid = threadIdx.x;
  __shared__ float red[256];
  const int N = NCLS * HID;
  float s = 0.0f;
  for (int k = tid; k < N; k += 256) s += fabsf(wfc[k]);
  red[tid] = s; __syncthreads();
  for (int off = 128; off > 0; off >>= 1) {
    if (tid < off) red[tid] += red[tid + off];
    __syncthreads();
  }
  if (tid == 0) scale_fc[0] = red[0] / (float)N;
  for (int k = tid; k < N; k += 256) wqs[k] = (wfc[k] >= 0.0f) ? 0x00 : 0x80;
}

__global__ __launch_bounds__(256)
void bnn_gemm_fp8(const uint8_t* __restrict__ A, const uint8_t* __restrict__ Wq,
                  const float* __restrict__ scale, const float* __restrict__ bias,
                  const float* __restrict__ gg, const float* __restrict__ bbe,
                  const float* __restrict__ mm, const float* __restrict__ vv,
                  uint8_t* __restrict__ Aout, int lda, int ksteps) {
  __shared__ uint8_t lds_a[128 * 128];

  const int tid  = threadIdx.x;
  const int wave = tid >> 5;
  const int lane = tid & 31;
  const int ln   = lane & 15;
  const int kh   = lane >> 4;
  const int m0   = blockIdx.y * 128;
  const int n0   = blockIdx.x * 128;
  const int n    = n0 + wave * 16 + ln;

  v8f acc[8] = {};

  const uint8_t* wrow = Wq + (size_t)n * lda;

  for (int ks = 0; ks < ksteps; ++ks) {
    const int k0 = ks * 128;
    __syncthreads();
#pragma unroll
    for (int i = 0; i < 4; ++i) {
      int q   = tid + i * 256;
      int r   = q >> 3;
      int off = (q & 7) * 16;
      const uint8_t* gsrc = A + (size_t)(m0 + r) * lda + k0 + off;
#if USE_ASYNC_LDS
      __builtin_amdgcn_global_load_async_to_lds_b128(
          (AS1 v4i*)gsrc, (AS3 v4i*)&lds_a[r * 128 + off], 0, 0);
#else
      *(v4i*)&lds_a[r * 128 + off] = *(const v4i*)gsrc;
#endif
    }
#if USE_ASYNC_LDS
    wait_asynccnt0();
#endif
    __syncthreads();

    v16i bt;
#pragma unroll
    for (int q = 0; q < 4; ++q) {
      v4i t = *(const v4i*)(wrow + k0 + q * 32 + kh * 16);
      bt[q * 4 + 0] = t[0]; bt[q * 4 + 1] = t[1];
      bt[q * 4 + 2] = t[2]; bt[q * 4 + 3] = t[3];
    }

#pragma unroll
    for (int mt = 0; mt < 8; ++mt) {
      v16i at;
      const int rl = mt * 16 + ln;
#pragma unroll
      for (int c = 0; c < 4; ++c) {
        v4i t = *(const v4i*)&lds_a[rl * 128 + c * 32 + kh * 16];
        at[c * 4 + 0] = t[0]; at[c * 4 + 1] = t[1];
        at[c * 4 + 2] = t[2]; at[c * 4 + 3] = t[3];
      }
      acc[mt] = __builtin_amdgcn_wmma_f32_16x16x128_fp8_fp8(
          at, bt, (short)0, acc[mt], false, false);
    }
  }

  const float sc  = scale[n];
  const float bb  = bias[n];
  const float rs  = rsqrtf(vv[n] + 1e-5f) * gg[n];
  const float mu  = mm[n];
  const float bet = bbe[n];
#pragma unroll
  for (int mt = 0; mt < 8; ++mt) {
    const int mbase = m0 + mt * 16 + kh * 8;
#pragma unroll
    for (int r = 0; r < 8; ++r) {
      float t = acc[mt][r] * sc + bb;
      t = (t - mu) * rs + bet;
      Aout[(size_t)(mbase + r) * HID + n] = (t >= 0.0f) ? 0x38 : 0xB8;
    }
  }
}

__global__ void bnn_fc(const uint8_t* __restrict__ A3, const uint8_t* __restrict__ wqs,
                       const float* __restrict__ scale_fc, const float* __restrict__ bfc,
                       float* __restrict__ out) {
  const int tid  = threadIdx.x;
  const int wave = tid >> 5;
  const int lane = tid & 31;
  const int row  = blockIdx.x * 8 + wave;
  if (lane >= NCLS) return;
  const uint32_t* ar = (const uint32_t*)(A3 + (size_t)row * HID);
  const uint32_t* wr = (const uint32_t*)(wqs + (size_t)lane * HID);
  int mis = 0;
#pragma unroll 8
  for (int i = 0; i < HID / 4; ++i)
    mis += __popc(((ar[i] ^ wr[i]) >> 7) & 0x01010101u);
  float s = (float)(HID - 2 * mis);
  out[(size_t)row * NCLS + lane] = s * scale_fc[0] + bfc[lane];
}

extern "C" void kernel_launch(void* const* d_in, const int* in_sizes, int n_in,
                              void* d_out, int out_size, void* d_ws, size_t ws_size,
                              hipStream_t stream) {
  (void)in_sizes; (void)n_in; (void)out_size; (void)ws_size;
  const float* x   = (const float*)d_in[0];
  const float* w1  = (const float*)d_in[1];
  const float* b1  = (const float*)d_in[2];
  const float* g1  = (const float*)d_in[3];
  const float* be1 = (const float*)d_in[4];
  const float* m1  = (const float*)d_in[5];
  const float* v1  = (const float*)d_in[6];
  const float* w2  = (const float*)d_in[7];
  const float* b2  = (const float*)d_in[8];
  const float* g2  = (const float*)d_in[9];
  const float* be2 = (const float*)d_in[10];
  const float* m2  = (const float*)d_in[11];
  const float* v2  = (const float*)d_in[12];
  const float* w3  = (const float*)d_in[13];
  const float* b3  = (const float*)d_in[14];
  const float* g3  = (const float*)d_in[15];
  const float* be3 = (const float*)d_in[16];
  const float* m3  = (const float*)d_in[17];
  const float* v3  = (const float*)d_in[18];
  const float* wfc = (const float*)d_in[19];
  const float* bfc = (const float*)d_in[20];
  float* out = (float*)d_out;

  uint8_t* ws = (uint8_t*)d_ws;
  size_t off = 0;
  auto carve = [&](size_t bytes) -> void* {
    void* p = ws + off;
    off = (off + bytes + 255) & ~(size_t)255;
    return p;
  };
  uint8_t* A0   = (uint8_t*)carve((size_t)BATCH * KP1);
  uint8_t* A1   = (uint8_t*)carve((size_t)BATCH * HID);
  uint8_t* A2   = (uint8_t*)carve((size_t)BATCH * HID);
  uint8_t* A3   = (uint8_t*)carve((size_t)BATCH * HID);
  uint8_t* Wq1  = (uint8_t*)carve((size_t)HID * KP1);
  uint8_t* Wq2  = (uint8_t*)carve((size_t)HID * HID);
  uint8_t* Wq3  = (uint8_t*)carve((size_t)HID * HID);
  uint8_t* WqF  = (uint8_t*)carve((size_t)NCLS * HID);
  float*   s1   = (float*)carve(HID * sizeof(float));
  float*   s2   = (float*)carve(HID * sizeof(float));
  float*   s3   = (float*)carve(HID * sizeof(float));
  float*   sfc  = (float*)carve(sizeof(float));

  bnn_quant_input<<<(BATCH * KP1) / 256, 256, 0, stream>>>(x, A0);
  bnn_quant_weight<<<HID, 256, 0, stream>>>(w1, Wq1, s1, DIN, KP1);
  bnn_quant_weight<<<HID, 256, 0, stream>>>(w2, Wq2, s2, HID, HID);
  bnn_quant_weight<<<HID, 256, 0, stream>>>(w3, Wq3, s3, HID, HID);
  bnn_quant_fc<<<1, 256, 0, stream>>>(wfc, WqF, sfc);

  dim3 grid(HID / 128, BATCH / 128);
  bnn_gemm_fp8<<<grid, 256, 0, stream>>>(A0, Wq1, s1, b1, g1, be1, m1, v1, A1, KP1, KP1 / 128);
  bnn_gemm_fp8<<<grid, 256, 0, stream>>>(A1, Wq2, s2, b2, g2, be2, m2, v2, A2, HID, HID / 128);
  bnn_gemm_fp8<<<grid, 256, 0, stream>>>(A2, Wq3, s3, b3, g3, be3, m3, v3, A3, HID, HID / 128);

  bnn_fc<<<BATCH / 8, 256, 0, stream>>>(A3, WqF, sfc, bfc, out);
}
